// LSTM_ATTACK_58789512347837
// MI455X (gfx1250) — matrix operand order repaired and checked by reading, not compiled, not run on hardware
//
#include <hip/hip_runtime.h>
#include <hip/hip_bf16.h>

typedef __attribute__((ext_vector_type(16))) _Float16 v16h;
typedef __attribute__((ext_vector_type(8)))  _Float16 v8h;
typedef __attribute__((ext_vector_type(8)))  float    v8f;
typedef __attribute__((ext_vector_type(4)))  int      v4i;

#define B_   512
#define T_   256
#define I_   256
#define H_   512
#define G4H  2048     // 4*H
#define KC   768      // H + I (recurrent + input K fused)
#define LDK  40       // LDS row stride in halves (keeps 16B alignment, skews banks)

// ---------------------------------------------------------------------------
// CDNA5 async global->LDS copy (ASYNCcnt path), with safe fallback.
// Probe-confirmed signature: (AS1 v4i* src, AS3 v4i* dst, imm off, imm cpol).
// ---------------------------------------------------------------------------
#if defined(__has_builtin)
#if __has_builtin(__builtin_amdgcn_global_load_async_to_lds_b128)
#define USE_ASYNC_LDS 1
#endif
#endif

#ifdef USE_ASYNC_LDS
typedef __attribute__((address_space(1))) v4i* gv4p;
typedef __attribute__((address_space(3))) v4i* lv4p;
#endif

__device__ __forceinline__ void cp16(const _Float16* g, _Float16* l) {
#ifdef USE_ASYNC_LDS
  __builtin_amdgcn_global_load_async_to_lds_b128(
      (gv4p)(void*)const_cast<_Float16*>(g), (lv4p)(void*)l, 0, 0);
#else
  *(v8h*)l = *(const v8h*)g;
#endif
}

__device__ __forceinline__ void cp_wait() {
#ifdef USE_ASYNC_LDS
#if __has_builtin(__builtin_amdgcn_s_wait_asynccnt)
  __builtin_amdgcn_s_wait_asynccnt(0);
#else
  asm volatile("s_wait_asynccnt 0" ::: "memory");
#endif
#endif
}

// ---------------------------------------------------------------------------
// WMMA fragment loaders (wave32, v_wmma_f32_16x16x32_f16 layouts per ISA 7.12.2)
// A 16x32 f16: lane l -> row m=l&15; halves 0..7 = K kh..kh+7, halves 8..15 =
//              K 16+kh..16+kh+7, kh = (l>>4)*8.
// B 32x16 f16 (stored transposed as Bt[n][k]): lane l -> col n=l&15; halves
//              0..7 = K ks..ks+7, halves 8..15 = K 16+ks..16+ks+7, ks = (l>>4)*8.
// ---------------------------------------------------------------------------
__device__ __forceinline__ v16h frag_a(const _Float16* base, int lane) {
  int m = lane & 15, kh = (lane >> 4) * 8;
  const _Float16* p = base + m * LDK + kh;
  v8h lo = *(const v8h*)p;
  v8h hi = *(const v8h*)(p + 16);
  v16h f;
#pragma unroll
  for (int i = 0; i < 8; ++i) { f[i] = lo[i]; f[8 + i] = hi[i]; }
  return f;
}

__device__ __forceinline__ v16h frag_b(const _Float16* base, int lane) {
  int n = lane & 15, ks = (lane >> 4) * 8;
  const _Float16* p = base + n * LDK + ks;
  v8h lo = *(const v8h*)p;
  v8h hi = *(const v8h*)(p + 16);
  v16h f;
#pragma unroll
  for (int i = 0; i < 8; ++i) { f[i] = lo[i]; f[8 + i] = hi[i]; }
  return f;
}

__device__ __forceinline__ float sigf(float x) { return 1.f / (1.f + __expf(-x)); }

// ---------------------------------------------------------------------------
// Prep kernels: one-time f32 -> f16 conversions / packing
// ---------------------------------------------------------------------------
__global__ __launch_bounds__(256) void k_prep_x(const float* __restrict__ x,
                                                _Float16* __restrict__ xT) {
  size_t idx = (size_t)blockIdx.x * 256 + threadIdx.x;   // over B*T*I
  int i = (int)(idx % I_);
  size_t bt = idx / I_;
  int t = (int)(bt % T_);
  int b = (int)(bt / T_);
  xT[((size_t)t * B_ + b) * I_ + i] = (_Float16)x[idx];  // [T, B, I] f16
}

__global__ __launch_bounds__(256) void k_prep_w(const float* __restrict__ Wih,
                                                const float* __restrict__ Whh,
                                                _Float16* __restrict__ Wc) {
  size_t idx = (size_t)blockIdx.x * 256 + threadIdx.x;   // over G4H*KC
  int k = (int)(idx % KC);
  int n = (int)(idx / KC);
  float v = (k < H_) ? Whh[(size_t)n * H_ + k] : Wih[(size_t)n * I_ + (k - H_)];
  Wc[idx] = (_Float16)v;                                  // [4H, H+I] f16
}

__global__ __launch_bounds__(256) void k_prep_misc(const float* __restrict__ bih,
                                                   const float* __restrict__ bhh,
                                                   const float* __restrict__ fc1w,
                                                   float* __restrict__ biasc,
                                                   _Float16* __restrict__ fc1wh,
                                                   _Float16* __restrict__ h0,
                                                   float* __restrict__ c0) {
  int idx = blockIdx.x * 256 + threadIdx.x;               // grid covers B_*H_
  if (idx < G4H)     biasc[idx] = bih[idx] + bhh[idx];
  if (idx < 32 * H_) fc1wh[idx] = (_Float16)fc1w[idx];
  if (idx < B_ * H_) { h0[idx] = (_Float16)0.f; c0[idx] = 0.f; }
}

// ---------------------------------------------------------------------------
// One LSTM time step: gates = [h_prev | x_t] (M=512,K=768) x Wc^T (N=2048),
// then fused cell update. Grid 128 WGs x 256 threads (8 waves).
// WG = 64 batch rows x 32 hidden cols x all 4 gates (128 N columns).
// Double-buffered async->LDS pipeline, one barrier per K-chunk.
// ---------------------------------------------------------------------------
__global__ __launch_bounds__(256)
void k_lstm_step(const _Float16* __restrict__ h_prev,
                 _Float16* __restrict__ h_next,
                 float* __restrict__ c_state,
                 const _Float16* __restrict__ xT,
                 const _Float16* __restrict__ Wc,
                 const float* __restrict__ biasc,
                 _Float16* __restrict__ lstm_out,
                 int t) {
  __shared__ alignas(16) _Float16 As[2][64 * LDK];
  __shared__ alignas(16) _Float16 Bs[2][128 * LDK];
  __shared__ float Gs[64 * 128];

  const int tid = threadIdx.x;
  const int lane = tid & 31, w = tid >> 5;
  const int mBase = (blockIdx.x >> 4) * 64;    // batch tile
  const int j0 = (blockIdx.x & 15) * 32;       // hidden tile
  const int mr = w >> 2, g = w & 3;

  // Per-thread tile-fill assignments (each thread: 1x b128 for A, 2x for B).
  const int arow = tid >> 2, apart = tid & 3;
  const int brow = tid >> 1, bpart = tid & 1;
  const int bng = (brow >> 5) * H_ + j0 + (brow & 31);   // global N row of Wc

  auto issue = [&](int kc) {
    const int buf = kc & 1;
    const int k0 = kc * 32;
    const _Float16* asrc =
        (k0 < H_) ? h_prev + (size_t)(mBase + arow) * H_ + k0 + apart * 8
                  : xT + ((size_t)t * B_ + (mBase + arow)) * I_ + (k0 - H_) + apart * 8;
    cp16(asrc, &As[buf][arow * LDK + apart * 8]);
    const _Float16* bsrc = Wc + (size_t)bng * KC + k0 + bpart * 16;
    cp16(bsrc,     &Bs[buf][brow * LDK + bpart * 16]);
    cp16(bsrc + 8, &Bs[buf][brow * LDK + bpart * 16 + 8]);
    if (kc + 2 < KC / 32)                      // keep next weight tile warm in L2
      __builtin_prefetch(Wc + (size_t)bng * KC + k0 + 64, 0, 1);
  };

  v8f acc[2][2] = {};
  issue(0);
  for (int kc = 0; kc < KC / 32; ++kc) {
    cp_wait();           // my async fills (incl. current buffer) complete
    __syncthreads();     // s_wait_dscnt 0 + barrier: tile visible, prev reads done
    if (kc + 1 < KC / 32) issue(kc + 1);

    const _Float16* a_ = As[kc & 1];
    const _Float16* b_ = Bs[kc & 1];
    v16h a0 = frag_a(a_ + (mr * 32 + 0) * LDK, lane);
    v16h a1 = frag_a(a_ + (mr * 32 + 16) * LDK, lane);
    v16h b0 = frag_b(b_ + (g * 32 + 0) * LDK, lane);
    v16h b1 = frag_b(b_ + (g * 32 + 16) * LDK, lane);
    acc[0][0] = __builtin_amdgcn_wmma_f32_16x16x32_f16(false, a0, false, b0, (short)0, acc[0][0], false, false);
    acc[0][1] = __builtin_amdgcn_wmma_f32_16x16x32_f16(false, a0, false, b1, (short)0, acc[0][1], false, false);
    acc[1][0] = __builtin_amdgcn_wmma_f32_16x16x32_f16(false, a1, false, b0, (short)0, acc[1][0], false, false);
    acc[1][1] = __builtin_amdgcn_wmma_f32_16x16x32_f16(false, a1, false, b1, (short)0, acc[1][1], false, false);
  }

  // Spill accumulators to LDS so the cell update sees all 4 gates per (b, j).
  {
    int hi = lane >> 4, n = lane & 15;
#pragma unroll
    for (int s = 0; s < 2; ++s)
#pragma unroll
      for (int s2 = 0; s2 < 2; ++s2)
#pragma unroll
        for (int r = 0; r < 8; ++r) {
          int M = mr * 32 + s * 16 + r + 8 * hi;
          int N = g * 32 + s2 * 16 + n;
          Gs[M * 128 + N] = acc[s][s2][r];
        }
  }
  __syncthreads();

  // Fused LSTM cell update: c' = f*c + i*g ; h = o*tanh(c')
  for (int item = tid; item < 64 * 32; item += 256) {
    int bl = item >> 5, jj = item & 31;
    int b = mBase + bl, j = j0 + jj;
    float xi = Gs[bl * 128 + jj]      + biasc[j];
    float xf = Gs[bl * 128 + 32 + jj] + biasc[H_ + j];
    float xg = Gs[bl * 128 + 64 + jj] + biasc[2 * H_ + j];
    float xo = Gs[bl * 128 + 96 + jj] + biasc[3 * H_ + j];
    float ii = sigf(xi), ff = sigf(xf), gg = tanhf(xg), oo = sigf(xo);
    size_t cidx = (size_t)b * H_ + j;
    float cn = ff * c_state[cidx] + ii * gg;
    c_state[cidx] = cn;
    float hh = oo * tanhf(cn);
    h_next[cidx] = (_Float16)hh;
    lstm_out[((size_t)b * T_ + t) * H_ + j] = (_Float16)hh;
  }
}

// ---------------------------------------------------------------------------
// fc1: a[M=131072, 32] = lstm_out[M, 512] x fc1_w^T + fc1_b   (WMMA)
// 4 waves/WG, WG tile 128x32, grid 1024. Async tile fill.
// ---------------------------------------------------------------------------
__global__ __launch_bounds__(128)
void k_fc1(const _Float16* __restrict__ A, const _Float16* __restrict__ Wf,
           const float* __restrict__ bias, float* __restrict__ out) {
  __shared__ alignas(16) _Float16 As[128 * LDK];
  __shared__ alignas(16) _Float16 Bs[32 * LDK];
  const int tid = threadIdx.x, lane = tid & 31, w = tid >> 5;
  const size_t mBase = (size_t)blockIdx.x * 128;

  v8f acc[2][2] = {};
  for (int kc = 0; kc < H_ / 32; ++kc) {
    const int k0 = kc * 32;
    { // A: each thread fills one 32-half row (4x b128)
      const _Float16* src = A + (mBase + tid) * H_ + k0;
#pragma unroll
      for (int p = 0; p < 4; ++p)
        cp16(src + p * 8, As + tid * LDK + p * 8);
    }
    { // B: Wf is [32, 512] row-major == B^T layout
      int r = tid >> 2, part = tid & 3;
      cp16(Wf + (size_t)r * H_ + k0 + part * 8, Bs + r * LDK + part * 8);
    }
    cp_wait();
    __syncthreads();
    v16h a0 = frag_a(As + (w * 32) * LDK, lane);
    v16h a1 = frag_a(As + (w * 32 + 16) * LDK, lane);
    v16h b0 = frag_b(Bs, lane);
    v16h b1 = frag_b(Bs + 16 * LDK, lane);
    acc[0][0] = __builtin_amdgcn_wmma_f32_16x16x32_f16(false, a0, false, b0, (short)0, acc[0][0], false, false);
    acc[0][1] = __builtin_amdgcn_wmma_f32_16x16x32_f16(false, a0, false, b1, (short)0, acc[0][1], false, false);
    acc[1][0] = __builtin_amdgcn_wmma_f32_16x16x32_f16(false, a1, false, b0, (short)0, acc[1][0], false, false);
    acc[1][1] = __builtin_amdgcn_wmma_f32_16x16x32_f16(false, a1, false, b1, (short)0, acc[1][1], false, false);
    __syncthreads();
  }
  int hi = lane >> 4, n = lane & 15;
#pragma unroll
  for (int s = 0; s < 2; ++s)
#pragma unroll
    for (int s2 = 0; s2 < 2; ++s2)
#pragma unroll
      for (int r = 0; r < 8; ++r) {
        int M = w * 32 + s * 16 + r + 8 * hi;
        int N = s2 * 16 + n;
        out[(mBase + M) * 32 + N] = acc[s][s2][r] + bias[N];
      }
}

// ---------------------------------------------------------------------------
// BatchNorm over channel t (torch BN1d semantics, biased var) + head
// ---------------------------------------------------------------------------
__global__ __launch_bounds__(256)
void k_bn1_stats(const float* __restrict__ a, float* __restrict__ mr) {
  const int t = blockIdx.x;
  __shared__ float ss[256], sq[256];
  float s = 0.f, q = 0.f;
  for (int i = threadIdx.x; i < B_ * 32; i += 256) {
    int b = i >> 5, o = i & 31;
    float x = a[((size_t)b * T_ + t) * 32 + o];
    s += x; q += x * x;
  }
  ss[threadIdx.x] = s; sq[threadIdx.x] = q;
  __syncthreads();
  for (int st = 128; st > 0; st >>= 1) {
    if (threadIdx.x < st) { ss[threadIdx.x] += ss[threadIdx.x + st]; sq[threadIdx.x] += sq[threadIdx.x + st]; }
    __syncthreads();
  }
  if (threadIdx.x == 0) {
    float m = ss[0] / (float)(B_ * 32);
    float var = sq[0] / (float)(B_ * 32) - m * m;
    mr[t] = m; mr[T_ + t] = rsqrtf(var + 1e-5f);
  }
}

__global__ __launch_bounds__(256)
void k_head_z(const float* __restrict__ a, const float* __restrict__ mr,
              const float* __restrict__ g1, const float* __restrict__ b1,
              const float* __restrict__ w2, const float* __restrict__ b2,
              float* __restrict__ z) {
  int idx = blockIdx.x * 256 + threadIdx.x;   // idx = b*T + t
  int t = idx % T_, b = idx / T_;
  float m = mr[t], r = mr[T_ + t], ga = g1[t], be = b1[t];
  float s = 0.f;
#pragma unroll
  for (int o = 0; o < 32; ++o) {
    float v = (a[(size_t)idx * 32 + o] - m) * r * ga + be;
    v = v < 0.f ? 0.1f * v : v;               // leaky_relu(0.1)
    s += v * w2[o];
  }
  z[(size_t)t * B_ + b] = s + b2[0];          // [T, B] for contiguous BN2 reduce
}

__global__ __launch_bounds__(256)
void k_bn2_stats(const float* __restrict__ z, float* __restrict__ mr) {
  const int t = blockIdx.x;
  __shared__ float ss[256], sq[256];
  float x0 = z[(size_t)t * B_ + threadIdx.x];
  float x1 = z[(size_t)t * B_ + threadIdx.x + 256];
  ss[threadIdx.x] = x0 + x1; sq[threadIdx.x] = x0 * x0 + x1 * x1;
  __syncthreads();
  for (int st = 128; st > 0; st >>= 1) {
    if (threadIdx.x < st) { ss[threadIdx.x] += ss[threadIdx.x + st]; sq[threadIdx.x] += sq[threadIdx.x + st]; }
    __syncthreads();
  }
  if (threadIdx.x == 0) {
    float m = ss[0] / (float)B_;
    float var = sq[0] / (float)B_ - m * m;
    mr[t] = m; mr[T_ + t] = rsqrtf(var + 1e-5f);
  }
}

__global__ __launch_bounds__(256)
void k_head_out(const float* __restrict__ z, const float* __restrict__ mr,
                const float* __restrict__ g2, const float* __restrict__ b2,
                float* __restrict__ out) {
  int idx = blockIdx.x * 256 + threadIdx.x;   // idx = b*T + t
  int t = idx % T_, b = idx / T_;
  float v = (z[(size_t)t * B_ + b] - mr[t]) * mr[T_ + t] * g2[t] + b2[t];
  out[idx] = v > 0.f ? v : 0.f;               // relu -> [B, T, 1]
}

// ---------------------------------------------------------------------------
extern "C" void kernel_launch(void* const* d_in, const int* in_sizes, int n_in,
                              void* d_out, int out_size, void* d_ws, size_t ws_size,
                              hipStream_t stream) {
  (void)in_sizes; (void)n_in; (void)out_size; (void)ws_size;
  const float* x     = (const float*)d_in[0];
  const float* Wih   = (const float*)d_in[1];
  const float* Whh   = (const float*)d_in[2];
  const float* bih   = (const float*)d_in[3];
  const float* bhh   = (const float*)d_in[4];
  const float* fc1_w = (const float*)d_in[5];
  const float* fc1_b = (const float*)d_in[6];
  const float* fc2_w = (const float*)d_in[7];
  const float* fc2_b = (const float*)d_in[8];
  const float* bn1_g = (const float*)d_in[9];
  const float* bn1_b = (const float*)d_in[10];
  const float* bn2_g = (const float*)d_in[11];
  const float* bn2_b = (const float*)d_in[12];

  char* p = (char*)d_ws;
  auto alloc = [&](size_t bytes) -> void* {
    void* r = (void*)p;
    p += (bytes + 255) & ~(size_t)255;
    return r;
  };
  _Float16* xT      = (_Float16*)alloc((size_t)T_ * B_ * I_ * 2);   //  67 MB
  _Float16* Wc      = (_Float16*)alloc((size_t)G4H * KC * 2);       //   3 MB
  float*    biasc   = (float*)   alloc((size_t)G4H * 4);
  _Float16* h0      = (_Float16*)alloc((size_t)B_ * H_ * 2);
  _Float16* h1      = (_Float16*)alloc((size_t)B_ * H_ * 2);
  float*    cst     = (float*)   alloc((size_t)B_ * H_ * 4);
  _Float16* lstmout = (_Float16*)alloc((size_t)B_ * T_ * H_ * 2);   // 134 MB
  _Float16* fc1wh   = (_Float16*)alloc((size_t)32 * H_ * 2);
  float*    abuf    = (float*)   alloc((size_t)B_ * T_ * 32 * 4);   //  17 MB
  float*    mr1     = (float*)   alloc((size_t)2 * T_ * 4);
  float*    zbuf    = (float*)   alloc((size_t)B_ * T_ * 4);
  float*    mr2     = (float*)   alloc((size_t)2 * T_ * 4);

  // One-time packing / f16 conversion
  k_prep_x<<<(B_ * T_ * I_) / 256, 256, 0, stream>>>(x, xT);
  k_prep_w<<<(G4H * KC) / 256, 256, 0, stream>>>(Wih, Whh, Wc);
  k_prep_misc<<<(B_ * H_) / 256, 256, 0, stream>>>(bih, bhh, fc1_w, biasc, fc1wh, h0, cst);

  // Sequential recurrence: 256 stream-ordered step GEMMs (f16 WMMA, f32 acc)
  for (int t = 0; t < T_; ++t) {
    const _Float16* hin = (t & 1) ? h1 : h0;
    _Float16* hout      = (t & 1) ? h0 : h1;
    k_lstm_step<<<128, 256, 0, stream>>>(hin, hout, cst, xT, Wc, biasc, lstmout, t);
  }

  // Head
  k_fc1<<<(B_ * T_) / 128, 128, 0, stream>>>(lstmout, fc1wh, fc1_b, abuf);
  k_bn1_stats<<<T_, 256, 0, stream>>>(abuf, mr1);
  k_head_z<<<(B_ * T_) / 256, 256, 0, stream>>>(abuf, mr1, bn1_g, bn1_b, fc2_w, fc2_b, zbuf);
  k_bn2_stats<<<T_, 256, 0, stream>>>(zbuf, mr2);
  k_head_out<<<(B_ * T_) / 256, 256, 0, stream>>>(zbuf, mr2, bn2_g, bn2_b, (float*)d_out);
}
